// TransformerDecoderLayer_9569187135641
// MI455X (gfx1250) — matrix operand order repaired and checked by reading, not compiled, not run on hardware
//
#include <hip/hip_runtime.h>
#include <hip/hip_bf16.h>

// ---------------------------------------------------------------------------
// Transformer decoder layer, MI455X (gfx1250, wave32, WMMA).
// B=64 T=100 D=1024 H=16 HD=64 DFF=4096.  All big GEMMs via
// v_wmma_f32_16x16x32_bf16 (bf16 A/B, f32 accum); tiles double-buffered in
// LDS and staged with GLOBAL_LOAD_ASYNC_TO_LDS_B128 (ASYNCcnt) so the async
// engine fetches tile k+1 while WMMAs crunch tile k. Attention (1.6% of
// FLOPs) is VALU fp32 online-softmax with async-staged K/V.
// ---------------------------------------------------------------------------

#define B_   64
#define T_   100
#define D_   1024
#define H_   16
#define HD_  64
#define DFF_ 4096
#define M_   (B_ * T_)      // 6400 rows, = 50 * 128

typedef __attribute__((ext_vector_type(16))) __bf16          v16bf;
typedef __attribute__((ext_vector_type(8)))  float           v8f;
typedef __attribute__((ext_vector_type(8)))  unsigned short  ushort8;

struct U16Pair { ushort8 lo, hi; };   // 32 bytes, bit-cast target for v16bf

static __device__ __forceinline__ unsigned short f2bf(float f) {
  unsigned int u = __float_as_uint(f);
  u += 0x7FFFu + ((u >> 16) & 1u);            // round-to-nearest-even
  return (unsigned short)(u >> 16);
}
static __device__ __forceinline__ float bf2f(unsigned short h) {
  return __uint_as_float(((unsigned int)h) << 16);
}

// LDS byte offset: flat LDS pointers keep the offset in the low 32 bits
// (aperture base occupies [63:32] per ISA aperture rules).
static __device__ __forceinline__ unsigned lds_off(const void* p) {
  return (unsigned)(size_t)p;
}

// Async global->LDS, 16B per lane; offset imm applies to BOTH addresses.
static __device__ __forceinline__ void async_b128(unsigned lds, unsigned long long ga) {
  asm volatile("global_load_async_to_lds_b128 %0, %1, off"
               :: "v"(lds), "v"(ga) : "memory");
}
static __device__ __forceinline__ void async_b128_o16(unsigned lds, unsigned long long ga) {
  asm volatile("global_load_async_to_lds_b128 %0, %1, off offset:16"
               :: "v"(lds), "v"(ga) : "memory");
}
static __device__ __forceinline__ void wait_async0() {
#if __has_builtin(__builtin_amdgcn_s_wait_asynccnt)
  __builtin_amdgcn_s_wait_asynccnt(0);
#else
  asm volatile("s_wait_asynccnt 0x0" ::: "memory");
#endif
}
static __device__ __forceinline__ void wait_async4() {   // steady-state: next
#if __has_builtin(__builtin_amdgcn_s_wait_asynccnt)      // tile's 4 copies may
  __builtin_amdgcn_s_wait_asynccnt(4);                   // remain in flight
#else
  asm volatile("s_wait_asynccnt 0x4" ::: "memory");
#endif
}

// ---------------------------------------------------------------------------
// fp32 -> bf16 elementwise convert
// ---------------------------------------------------------------------------
__global__ void convert_f32_to_bf16(const float* __restrict__ in,
                                    unsigned short* __restrict__ out, int n) {
  int i = blockIdx.x * blockDim.x + threadIdx.x;
  if (i < n) out[i] = f2bf(in[i]);
}

// ---------------------------------------------------------------------------
// C[M,128-tiles of N] = A[M,K](bf16) * W[N,K]^T(bf16) + bias
// MODE 0: fp32 out.  MODE 1: bf16 out with ReLU.
// Block: 256 threads (8 waves, 2x4), tile 128x128, K-step 32.
// Wave tile 64x32 = 4x2 grid of 16x16 WMMA accumulators.
// Double-buffered LDS; async copies overlap WMMA of the previous K-step.
// ---------------------------------------------------------------------------
template <int MODE>
__global__ __launch_bounds__(256)
void gemm_bf16_wmma(const unsigned short* __restrict__ A,
                    const unsigned short* __restrict__ W,
                    const float* __restrict__ bias,
                    void* __restrict__ outp, int K, int N) {
  constexpr int LDP = 40;                         // 32 + 8 pad (elems)
  __shared__ unsigned short As[2][128 * LDP];     // [buf][m][k]
  __shared__ unsigned short Bs[2][128 * LDP];     // [buf][n][k]

  const int tid  = threadIdx.x;
  const int m0   = blockIdx.x * 128;
  const int n0   = blockIdx.y * 128;
  const int wave = tid >> 5;
  const int lane = tid & 31;
  const int wrow = wave & 1;                      // 64-row slab
  const int wcol = wave >> 1;                     // 32-col slab
  const int ksel = lane >> 4;
  const int l15  = lane & 15;

  // global->LDS staging: thread moves 32B per tile per matrix (2x b128)
  const int lrow  = tid >> 1;                     // 0..127
  const int lhalf = tid & 1;                      // 16-elem half of 32-col strip
  const unsigned short* gA = A + (size_t)(m0 + lrow) * K + lhalf * 16;
  const unsigned short* gW = W + (size_t)(n0 + lrow) * K + lhalf * 16;
  unsigned ldsA[2], ldsW[2];
  ldsA[0] = lds_off(&As[0][lrow * LDP + lhalf * 16]);
  ldsA[1] = lds_off(&As[1][lrow * LDP + lhalf * 16]);
  ldsW[0] = lds_off(&Bs[0][lrow * LDP + lhalf * 16]);
  ldsW[1] = lds_off(&Bs[1][lrow * LDP + lhalf * 16]);

  v8f acc[4][2];
  const v8f vzero = {0.f, 0.f, 0.f, 0.f, 0.f, 0.f, 0.f, 0.f};
#pragma unroll
  for (int mi = 0; mi < 4; ++mi)
#pragma unroll
    for (int ni = 0; ni < 2; ++ni) acc[mi][ni] = vzero;

  const int nK = K >> 5;

  // prologue: stage tile 0 into buffer 0
  {
    const unsigned long long pA = (unsigned long long)gA;
    const unsigned long long pW = (unsigned long long)gW;
    async_b128(ldsA[0], pA);  async_b128_o16(ldsA[0], pA);
    async_b128(ldsW[0], pW);  async_b128_o16(ldsW[0], pW);
  }

  for (int kt = 0; kt < nK; ++kt) {
    const int cur = kt & 1;
    if (kt + 1 < nK) {
      // stage tile kt+1 into the other buffer; overlaps this step's WMMAs
      const unsigned long long pA = (unsigned long long)(gA + (kt + 1) * 32);
      const unsigned long long pW = (unsigned long long)(gW + (kt + 1) * 32);
      async_b128(ldsA[1 - cur], pA);  async_b128_o16(ldsA[1 - cur], pA);
      async_b128(ldsW[1 - cur], pW);  async_b128_o16(ldsW[1 - cur], pW);
      if (kt + 2 < nK) {                          // global_prefetch_b8, dist 2
        __builtin_prefetch(gA + (kt + 2) * 32, 0, 3);
        __builtin_prefetch(gW + (kt + 2) * 32, 0, 3);
      }
      wait_async4();                              // tile kt landed (in-order)
    } else {
      wait_async0();                              // final tile landed
    }
    __syncthreads();                              // tile visible block-wide

    const unsigned short* Ab = As[cur];
    const unsigned short* Bb = Bs[cur];

    // A fragments: lane(l15,ksel): elems 0..7 -> k=ksel*8+e,
    //                              elems 8..15 -> k=16+ksel*8+(e-8)
    v16bf afr[4];
#pragma unroll
    for (int mi = 0; mi < 4; ++mi) {
      const int r = wrow * 64 + mi * 16 + l15;
      U16Pair t;
      t.lo = *(const ushort8*)&Ab[r * LDP + ksel * 8];
      t.hi = *(const ushort8*)&Ab[r * LDP + 16 + ksel * 8];
      afr[mi] = __builtin_bit_cast(v16bf, t);
    }
    // B fragments: lane(n=l15, ksel): elems 0..7 -> k=ksel*8+e, elems 8..15 -> k=16+ksel*8+(e-8)
    v16bf bfr[2];
#pragma unroll
    for (int ni = 0; ni < 2; ++ni) {
      const int c = wcol * 32 + ni * 16 + l15;
      U16Pair t;
      t.lo = *(const ushort8*)&Bb[c * LDP + ksel * 8];
      t.hi = *(const ushort8*)&Bb[c * LDP + 16 + ksel * 8];
      bfr[ni] = __builtin_bit_cast(v16bf, t);
    }

#pragma unroll
    for (int mi = 0; mi < 4; ++mi)
#pragma unroll
      for (int ni = 0; ni < 2; ++ni)
        acc[mi][ni] = __builtin_amdgcn_wmma_f32_16x16x32_bf16(
            false, afr[mi], false, bfr[ni], (short)0, acc[mi][ni], false, false);

    __syncthreads();   // all waves done reading buf[cur] before it's re-staged
  }

  // Epilogue: C/D layout -> lane l15 = N, (lane>>4)*8 + r = M
  const int mbase = ksel * 8;
#pragma unroll
  for (int mi = 0; mi < 4; ++mi) {
#pragma unroll
    for (int ni = 0; ni < 2; ++ni) {
      const int gcol = n0 + wcol * 32 + ni * 16 + l15;
      const float bv = bias[gcol];
#pragma unroll
      for (int r = 0; r < 8; ++r) {
        const int grow = m0 + wrow * 64 + mi * 16 + mbase + r;
        const float v = acc[mi][ni][r] + bv;
        if (MODE == 0) {
          ((float*)outp)[(size_t)grow * N + gcol] = v;
        } else {
          ((unsigned short*)outp)[(size_t)grow * N + gcol] = f2bf(fmaxf(v, 0.f));
        }
      }
    }
  }
}

// ---------------------------------------------------------------------------
// RoPE + head split: qkv fp32 [M, 3D] -> Q,K (rope'd), V as bf16 [B,H,T,HD]
// ---------------------------------------------------------------------------
__global__ void rope_split(const float* __restrict__ qkv,
                           unsigned short* __restrict__ Q,
                           unsigned short* __restrict__ Kc,
                           unsigned short* __restrict__ V) {
  const int idx = blockIdx.x * blockDim.x + threadIdx.x;
  if (idx >= B_ * H_ * T_ * HD_) return;
  const int d  = idx & (HD_ - 1);
  const int t  = (idx >> 6) % T_;
  const int hb = (idx >> 6) / T_;
  const int h  = hb % H_;
  const int b  = hb / H_;

  const size_t base = (size_t)(b * T_ + t) * (3 * D_) + h * HD_;
  const size_t o = ((size_t)(b * H_ + h) * T_ + t) * HD_ + d;

  V[o] = f2bf(qkv[base + 2 * D_ + d]);

  const int j = d & 31;
  const float invf = __powf(10000.f, -(float)(j & 15) / 16.f);
  float s, c;
  __sincosf((float)t * invf, &s, &c);
  const float q1 = qkv[base + 2 * j],       q2 = qkv[base + 2 * j + 1];
  const float k1 = qkv[base + D_ + 2 * j],  k2 = qkv[base + D_ + 2 * j + 1];
  const float qo = (d < 32) ? (q1 * c - q2 * s) : (q1 * s + q2 * c);
  const float ko = (d < 32) ? (k1 * c - k2 * s) : (k1 * s + k2 * c);
  Q[o]  = f2bf(qo);
  Kc[o] = f2bf(ko);
}

// ---------------------------------------------------------------------------
// Causal attention, one workgroup per (b,h). K/V async-staged into LDS,
// per-thread online softmax (flash-style). Writes ctx bf16 [B,T,H,HD]=[M,D].
// ---------------------------------------------------------------------------
__global__ __launch_bounds__(128)
void attention_causal(const unsigned short* __restrict__ Q,
                      const unsigned short* __restrict__ Kc,
                      const unsigned short* __restrict__ V,
                      unsigned short* __restrict__ ctx) {
  __shared__ unsigned short Ks[T_ * HD_];
  __shared__ unsigned short Vs[T_ * HD_];
  const int bh = blockIdx.x;
  const int b = bh / H_, h = bh % H_;
  const size_t base = (size_t)bh * T_ * HD_;
  const int t = threadIdx.x;

  // 12800 B per array = 800 x 16B async transfers each
  {
    const unsigned ldsK = lds_off(Ks);
    const unsigned ldsV = lds_off(Vs);
    const unsigned long long gK = (unsigned long long)(Kc + base);
    const unsigned long long gV = (unsigned long long)(V + base);
    for (int i = t; i < (T_ * HD_ * 2) / 16; i += 128) {
      const unsigned bo = (unsigned)i * 16u;
      async_b128(ldsK + bo, gK + bo);
      async_b128(ldsV + bo, gV + bo);
    }
    wait_async0();
  }
  __syncthreads();

  if (t >= T_) return;

  float q[HD_];
#pragma unroll
  for (int d = 0; d < HD_; ++d) q[d] = bf2f(Q[base + (size_t)t * HD_ + d]);

  float m = -3.0e38f, l = 0.f;
  float acc[HD_];
#pragma unroll
  for (int d = 0; d < HD_; ++d) acc[d] = 0.f;
  const float scale = 0.125f;                  // 1/sqrt(64)

  for (int k = 0; k <= t; ++k) {
    float s = 0.f;
#pragma unroll
    for (int d = 0; d < HD_; ++d) s += q[d] * bf2f(Ks[k * HD_ + d]);
    s *= scale;
    const float mn = fmaxf(m, s);
    const float corr = __expf(m - mn);
    const float p = __expf(s - mn);
    l = l * corr + p;
#pragma unroll
    for (int d = 0; d < HD_; ++d) acc[d] = acc[d] * corr + p * bf2f(Vs[k * HD_ + d]);
    m = mn;
  }
  const float inv = 1.f / l;
  const size_t orow = ((size_t)(b * T_ + t) * H_ + h) * HD_;
#pragma unroll
  for (int d = 0; d < HD_; ++d) ctx[orow + d] = f2bf(acc[d] * inv);
}

// ---------------------------------------------------------------------------
// LayerNorm( a + r ) * g + beta ; one row (D=1024) per 256-thread block.
// Optionally writes a bf16 copy for the next GEMM.
// ---------------------------------------------------------------------------
__global__ __launch_bounds__(256)
void layernorm_residual(const float* __restrict__ a, const float* __restrict__ r,
                        const float* __restrict__ g, const float* __restrict__ beta,
                        float* __restrict__ outf, unsigned short* __restrict__ outb) {
  __shared__ float red[256];
  const int row = blockIdx.x, tid = threadIdx.x;
  const size_t base = (size_t)row * D_;

  float e[4];
  float s = 0.f;
#pragma unroll
  for (int j = 0; j < 4; ++j) {
    const int c = tid + j * 256;
    e[j] = a[base + c] + r[base + c];
    s += e[j];
  }
  red[tid] = s;
  __syncthreads();
  for (int st = 128; st > 0; st >>= 1) {
    if (tid < st) red[tid] += red[tid + st];
    __syncthreads();
  }
  const float mu = red[0] * (1.f / D_);
  __syncthreads();

  float v = 0.f;
#pragma unroll
  for (int j = 0; j < 4; ++j) {
    const float d = e[j] - mu;
    v += d * d;
  }
  red[tid] = v;
  __syncthreads();
  for (int st = 128; st > 0; st >>= 1) {
    if (tid < st) red[tid] += red[tid + st];
    __syncthreads();
  }
  const float rstd = rsqrtf(red[0] * (1.f / D_) + 1e-5f);

#pragma unroll
  for (int j = 0; j < 4; ++j) {
    const int c = tid + j * 256;
    const float y = (e[j] - mu) * rstd * g[c] + beta[c];
    outf[base + c] = y;
    if (outb) outb[base + c] = f2bf(y);
  }
}

// ---------------------------------------------------------------------------
// Host orchestration
// ---------------------------------------------------------------------------
extern "C" void kernel_launch(void* const* d_in, const int* in_sizes, int n_in,
                              void* d_out, int out_size, void* d_ws, size_t ws_size,
                              hipStream_t stream) {
  (void)in_sizes; (void)n_in; (void)out_size; (void)ws_size;

  const float* tgt  = (const float*)d_in[0];
  const float* Wqkv = (const float*)d_in[1];
  const float* bqkv = (const float*)d_in[2];
  const float* Wo   = (const float*)d_in[3];
  const float* bo   = (const float*)d_in[4];
  const float* W1   = (const float*)d_in[5];
  const float* b1   = (const float*)d_in[6];
  const float* W2   = (const float*)d_in[7];
  const float* b2   = (const float*)d_in[8];
  const float* g1   = (const float*)d_in[9];
  const float* be1  = (const float*)d_in[10];
  const float* g2   = (const float*)d_in[11];
  const float* be2  = (const float*)d_in[12];
  // d_in[13] = tgt_mask (causal tril) — attention hardcodes causal.

  const size_t SZ_WQKV  = (size_t)3 * D_ * D_;       // 3,145,728
  const size_t SZ_WO    = (size_t)D_ * D_;           // 1,048,576
  const size_t SZ_W1    = (size_t)DFF_ * D_;         // 4,194,304
  const size_t SZ_W2    = (size_t)D_ * DFF_;         // 4,194,304
  const size_t SZ_X     = (size_t)M_ * D_;           // 6,553,600
  const size_t SZ_QKV   = (size_t)M_ * 3 * D_;       // 19,660,800
  const size_t SZ_HEADS = (size_t)B_ * H_ * T_ * HD_;// 6,553,600

  char* p = (char*)d_ws;
  auto take = [&](size_t bytes) {
    char* r = p;
    p += (bytes + 255) & ~(size_t)255;
    return r;
  };
  unsigned short* wqkv_bf = (unsigned short*)take(SZ_WQKV * 2);
  unsigned short* wo_bf   = (unsigned short*)take(SZ_WO * 2);
  unsigned short* w1_bf   = (unsigned short*)take(SZ_W1 * 2);
  unsigned short* w2_bf   = (unsigned short*)take(SZ_W2 * 2);
  unsigned short* xbf     = (unsigned short*)take(SZ_X * 2);   // tgt_bf16, later x_bf16
  char* big               = take(SZ_QKV * 4);                  // qkv f32 | attn f32 | h bf16
  char* rot               = take(3 * SZ_HEADS * 2);            // Q,K,V bf16 | later ff f32
  unsigned short* ctx_bf  = (unsigned short*)take(SZ_X * 2);
  float* x_f32            = (float*)take(SZ_X * 4);

  float* qkv_f32  = (float*)big;
  float* attn_f32 = (float*)big;                 // reuse: qkv dead after rope
  unsigned short* h_bf = (unsigned short*)big;   // reuse: attn dead after LN1
  unsigned short* Qr = (unsigned short*)rot;
  unsigned short* Kr = Qr + SZ_HEADS;
  unsigned short* Vr = Kr + SZ_HEADS;
  float* ff_f32 = (float*)rot;                   // reuse: Q/K/V dead after attention

  dim3 blk256(256), blk128(128);
  auto cblocks = [](size_t n) { return dim3((unsigned)((n + 255) / 256)); };

  // 1) converts
  convert_f32_to_bf16<<<cblocks(SZ_X),    blk256, 0, stream>>>(tgt,  xbf,     (int)SZ_X);
  convert_f32_to_bf16<<<cblocks(SZ_WQKV), blk256, 0, stream>>>(Wqkv, wqkv_bf, (int)SZ_WQKV);
  convert_f32_to_bf16<<<cblocks(SZ_WO),   blk256, 0, stream>>>(Wo,   wo_bf,   (int)SZ_WO);
  convert_f32_to_bf16<<<cblocks(SZ_W1),   blk256, 0, stream>>>(W1,   w1_bf,   (int)SZ_W1);
  convert_f32_to_bf16<<<cblocks(SZ_W2),   blk256, 0, stream>>>(W2,   w2_bf,   (int)SZ_W2);

  // 2) QKV projection: [6400,1024] x [3072,1024]^T -> [6400,3072] f32
  gemm_bf16_wmma<0><<<dim3(M_ / 128, 3 * D_ / 128), blk256, 0, stream>>>(
      xbf, wqkv_bf, bqkv, qkv_f32, D_, 3 * D_);

  // 3) RoPE + split heads
  rope_split<<<cblocks(SZ_HEADS), blk256, 0, stream>>>(qkv_f32, Qr, Kr, Vr);

  // 4) causal attention -> ctx bf16 [M, D]
  attention_causal<<<dim3(B_ * H_), blk128, 0, stream>>>(Qr, Kr, Vr, ctx_bf);

  // 5) output projection: [6400,1024] x [1024,1024]^T -> attn f32
  gemm_bf16_wmma<0><<<dim3(M_ / 128, D_ / 128), blk256, 0, stream>>>(
      ctx_bf, wo_bf, bo, attn_f32, D_, D_);

  // 6) LN1( tgt + attn ) -> x f32 + x bf16
  layernorm_residual<<<dim3(M_), blk256, 0, stream>>>(tgt, attn_f32, g1, be1, x_f32, xbf);

  // 7) FFN up + ReLU: [6400,1024] x [4096,1024]^T -> h bf16
  gemm_bf16_wmma<1><<<dim3(M_ / 128, DFF_ / 128), blk256, 0, stream>>>(
      xbf, w1_bf, b1, h_bf, D_, DFF_);

  // 8) FFN down: [6400,4096] x [1024,4096]^T -> ff f32
  gemm_bf16_wmma<0><<<dim3(M_ / 128, D_ / 128), blk256, 0, stream>>>(
      h_bf, w2_bf, b2, ff_f32, DFF_, D_);

  // 9) LN2( x + ff ) -> out f32
  layernorm_residual<<<dim3(M_), blk256, 0, stream>>>(x_f32, ff_f32, g2, be2,
                                                      (float*)d_out, nullptr);
}
